// CVAE_15144054686025
// MI455X (gfx1250) — matrix operand order repaired and checked by reading, not compiled, not run on hardware
//
#include <hip/hip_runtime.h>
#include <hip/hip_bf16.h>
#include <cstdint>

#define BB 64
#define TT 64
#define HH 1024
#define VV 32000
#define CC 10
#define ZZ 32
#define CDD 8
#define GG (4*HH)

typedef __attribute__((ext_vector_type(16))) __bf16 v16bf;
typedef __attribute__((ext_vector_type(8)))  float  v8f;

union FragBF { v16bf v; uint4 q[2]; };

__device__ __forceinline__ uint16_t f2bf(float f) {
    uint32_t u = __float_as_uint(f);
    uint32_t r = (u + 0x7FFFu + ((u >> 16) & 1u)) >> 16;
    return (uint16_t)r;
}

__device__ __forceinline__ void store4bf(uint16_t* p, float4 v) {
    uint2 o;
    o.x = (uint32_t)f2bf(v.x) | ((uint32_t)f2bf(v.y) << 16);
    o.y = (uint32_t)f2bf(v.z) | ((uint32_t)f2bf(v.w) << 16);
    *reinterpret_cast<uint2*>(p) = o;
}

__device__ __forceinline__ float sigf(float x) { return 1.0f / (1.0f + __expf(-x)); }

__device__ __forceinline__ void wait_async0() {
#if __has_builtin(__builtin_amdgcn_s_wait_asynccnt)
    __builtin_amdgcn_s_wait_asynccnt(0);
#else
    asm volatile("s_wait_asynccnt 0" ::: "memory");
#endif
}

// ---------------------------------------------------------------------------
// Wave-level GEMM: one wave computes a 16(M) x 64(N) f32 tile.
// A: bf16 row-major, pre-offset to (m0, k=0); Bm: bf16 row-major [N, ldb]
// (B^T rows), pre-offset to (n0, k=0).  D[m][n] += sum_k A[m][k]*Bm[n][k]
// ---------------------------------------------------------------------------
__device__ __forceinline__ void wave_gemm_16x64(const uint16_t* __restrict__ A,
                                                const uint16_t* __restrict__ Bm,
                                                int lda, int ldb, int K,
                                                v8f acc[4], int lane)
{
    const int mh = lane & 15, half = lane >> 4;
    const uint16_t* ar  = A  + mh * lda;
    const uint16_t* br0 = Bm + mh * ldb;
    for (int k0 = 0; k0 < K; k0 += 32) {
        FragBF a;
        a.q[0] = *reinterpret_cast<const uint4*>(ar + k0 + half * 8);
        a.q[1] = *reinterpret_cast<const uint4*>(ar + k0 + 16 + half * 8);
#pragma unroll
        for (int j = 0; j < 4; ++j) {
            const uint16_t* bp = br0 + j * 16 * ldb + k0 + half * 8;
            FragBF b;
            b.q[0] = *reinterpret_cast<const uint4*>(bp);
            b.q[1] = *reinterpret_cast<const uint4*>(bp + 16);
            acc[j] = __builtin_amdgcn_wmma_f32_16x16x32_bf16(
                false, a.v, false, b.v, (short)0, acc[j], false, false);
        }
    }
}

// ---------------------------------------------------------------------------
// f32 -> bf16 bulk convert (weights)
// ---------------------------------------------------------------------------
__global__ void cvt_bf16_kernel(const float* __restrict__ src, uint16_t* __restrict__ dst, int n4) {
    int i = blockIdx.x * blockDim.x + threadIdx.x;
    if (i < n4) {
        float4 v = reinterpret_cast<const float4*>(src)[i];
        store4bf(dst + (size_t)i * 4, v);
    }
}

// ---------------------------------------------------------------------------
// Embedding gather for encoder + decoder (time-major [T*B, H], bf16)
// ---------------------------------------------------------------------------
__global__ void embed_kernel(const int* __restrict__ words,
                             const float* __restrict__ embN,
                             const float* __restrict__ embD,
                             uint16_t* __restrict__ xe, uint16_t* __restrict__ xd) {
    int row = blockIdx.x;              // t*B + b
    int t = row >> 6, b = row & 63;
    int tokE = words[b * TT + t];
    int tokD = (t == 0) ? 0 : words[b * TT + t - 1];
    int j = threadIdx.x * 4;
    float4 ve = *reinterpret_cast<const float4*>(embN + (size_t)tokE * HH + j);
    float4 vd = *reinterpret_cast<const float4*>(embD + (size_t)tokD * HH + j);
    store4bf(xe + (size_t)row * HH + j, ve);
    store4bf(xd + (size_t)row * HH + j, vd);
}

// ---------------------------------------------------------------------------
// Encoder init: h0 = [zeros(H-CD) ; cond_e], c0 = 0; also saves cond_e
// ---------------------------------------------------------------------------
__global__ void enc_init_kernel(const int* __restrict__ cond, const float* __restrict__ emb_cond,
                                float* __restrict__ c, float* __restrict__ h_f32,
                                uint16_t* __restrict__ h_bf, float* __restrict__ cond_e) {
    int idx = blockIdx.x * blockDim.x + threadIdx.x;   // B*H
    int b = idx >> 10, j = idx & 1023;
    float v = 0.0f;
    if (j >= HH - CDD) v = emb_cond[cond[b] * CDD + (j - (HH - CDD))];
    c[idx] = 0.0f;
    h_f32[idx] = v;
    h_bf[idx] = f2bf(v);
    if (j < CDD) cond_e[b * CDD + j] = emb_cond[cond[b] * CDD + j];
}

// ---------------------------------------------------------------------------
// Fused LSTM step: gates = x_t@Wih^T + h@Whh^T + bih + bhh, then pointwise.
// Grid: H/64 blocks of 512 threads (16 waves). Wave w: gate q=w&3, m-tile w>>2.
// ---------------------------------------------------------------------------
__global__ __launch_bounds__(512)
void lstm_step_kernel(const uint16_t* __restrict__ x_t,   // [B,H] bf16
                      const uint16_t* __restrict__ h_in,  // [B,H] bf16
                      const uint16_t* __restrict__ Wih,   // [4H,H] bf16
                      const uint16_t* __restrict__ Whh,   // [4H,H] bf16
                      const float* __restrict__ bih, const float* __restrict__ bhh,
                      float* __restrict__ c,              // [B,H]
                      uint16_t* __restrict__ h_out,       // [B,H] bf16
                      float* __restrict__ h_f32,          // [B,H]
                      uint16_t* __restrict__ hs_out)      // [B,H] slice or null
{
    __shared__ float g_lds[4][64][64];                    // 64 KB
    const int tid = threadIdx.x, lane = tid & 31, w = tid >> 5;
    const int q = w & 3, mt = w >> 2;
    const int j0 = blockIdx.x * 64;
    const int m0 = mt * 16;
    const int n0 = q * HH + j0;

    v8f acc[4];
#pragma unroll
    for (int j = 0; j < 4; ++j) acc[j] = (v8f){0.f,0.f,0.f,0.f,0.f,0.f,0.f,0.f};

    wave_gemm_16x64(x_t  + m0 * HH, Wih + (size_t)n0 * HH, HH, HH, HH, acc, lane);
    wave_gemm_16x64(h_in + m0 * HH, Whh + (size_t)n0 * HH, HH, HH, HH, acc, lane);

    const int mh = lane & 15, half = lane >> 4;
#pragma unroll
    for (int j = 0; j < 4; ++j) {
        int n = n0 + j * 16 + mh;
        float bias = bih[n] + bhh[n];
#pragma unroll
        for (int r = 0; r < 8; ++r) {
            int m = m0 + half * 8 + r;
            g_lds[q][m][j * 16 + mh] = acc[j][r] + bias;
        }
    }
    __syncthreads();

    for (int e = tid; e < 64 * 64; e += 512) {
        int m = e >> 6, j = e & 63;
        float ig = sigf(g_lds[0][m][j]);
        float fg = sigf(g_lds[1][m][j]);
        float gg = tanhf(g_lds[2][m][j]);
        float og = sigf(g_lds[3][m][j]);
        int idx = m * HH + j0 + j;
        float cn = fg * c[idx] + ig * gg;
        float hn = og * tanhf(cn);
        c[idx] = cn;
        uint16_t hb = f2bf(hn);
        h_out[idx] = hb;
        h_f32[idx] = hn;
        if (hs_out) hs_out[idx] = hb;
    }
}

// ---------------------------------------------------------------------------
// latent = eps*exp(0.5*logvar) + mean ; cat with cond_e -> lat [B, Z+CD]
// ---------------------------------------------------------------------------
__global__ void latent_kernel(const float* __restrict__ hT,
                              const float* __restrict__ Wm, const float* __restrict__ bm,
                              const float* __restrict__ Wl, const float* __restrict__ bl,
                              const float* __restrict__ eps, const float* __restrict__ cond_e,
                              float* __restrict__ lat) {
    int i = blockIdx.x * blockDim.x + threadIdx.x;     // B*Z
    if (i >= BB * ZZ) return;
    int b = i >> 5, z = i & 31;
    const float* h = hT + b * HH;
    float sm = bm[z], sl = bl[z];
    for (int k = 0; k < HH; ++k) {
        float hv = h[k];
        sm += hv * Wm[z * HH + k];
        sl += hv * Wl[z * HH + k];
    }
    float lv = eps[b * ZZ + z] * __expf(0.5f * sl) + sm;
    lat[b * (ZZ + CDD) + z] = lv;
    if (z < CDD) lat[b * (ZZ + CDD) + ZZ + z] = cond_e[b * CDD + z];
}

// ---------------------------------------------------------------------------
// hd0 = lat @ W_st^T + b_st ; c0 = 0
// ---------------------------------------------------------------------------
__global__ void dec_init_kernel(const float* __restrict__ lat,
                                const float* __restrict__ Wst, const float* __restrict__ bst,
                                float* __restrict__ c, float* __restrict__ h_f32,
                                uint16_t* __restrict__ h_bf) {
    int idx = blockIdx.x * blockDim.x + threadIdx.x;   // B*H
    int b = idx >> 10, j = idx & 1023;
    float s = bst[j];
    const float* lb = lat + b * (ZZ + CDD);
#pragma unroll
    for (int k = 0; k < ZZ + CDD; ++k) s += lb[k] * Wst[j * (ZZ + CDD) + k];
    c[idx] = 0.0f;
    h_f32[idx] = s;
    h_bf[idx] = f2bf(s);
}

// ---------------------------------------------------------------------------
// logits = hs_d @ W_out^T + b_out, stored [B,T,V] f32.
// Block = 128 threads (4 waves), block tile 32(M) x 256(N); wave tile 32x64.
// A tile (32 rows x 1024 k = 64KB, contiguous) is staged into LDS with
// global_load_async_to_lds_b128 (ASYNCcnt) and shared by all 4 waves.
// Grid (V/256, T*B/32).
// ---------------------------------------------------------------------------
__global__ __launch_bounds__(128)
void out_gemm_kernel(const uint16_t* __restrict__ Abf,   // [T*B, H] bf16
                     const uint16_t* __restrict__ Wb,    // [V, H] bf16
                     const float* __restrict__ bout,
                     float* __restrict__ out) {
    __shared__ uint16_t aTile[32 * HH];                  // 64 KB
    const int tid = threadIdx.x;
    const int lane = tid & 31, w = tid >> 5;
    const int m0 = blockIdx.y * 32;
    const int n0 = blockIdx.x * 256 + w * 64;

    // ---- async copy of A tile (64KB contiguous) into LDS ----
    {
        const char* gsrc = (const char*)(Abf + (size_t)m0 * HH);
        char* ldst = (char*)aTile;
#pragma unroll 4
        for (int i = 0; i < 32; ++i) {
            int off = (i * 128 + tid) * 16;
            uint64_t ga = (uint64_t)(uintptr_t)(gsrc + off);
            uint32_t la = (uint32_t)(uintptr_t)(ldst + off);
            asm volatile("global_load_async_to_lds_b128 %0, %1, off"
                         :: "v"(la), "v"(ga) : "memory");
        }
        wait_async0();
        __syncthreads();
    }

    v8f acc[2][4];
#pragma unroll
    for (int s = 0; s < 2; ++s)
#pragma unroll
        for (int j = 0; j < 4; ++j)
            acc[s][j] = (v8f){0.f,0.f,0.f,0.f,0.f,0.f,0.f,0.f};

    const int mh = lane & 15, half = lane >> 4;
    const uint16_t* ar0 = aTile + mh * HH;               // m-subtile 0
    const uint16_t* ar1 = aTile + (16 + mh) * HH;        // m-subtile 1
    const uint16_t* br0 = Wb + (size_t)(n0 + mh) * HH;

    for (int k0 = 0; k0 < HH; k0 += 32) {
        FragBF a0, a1;
        a0.q[0] = *reinterpret_cast<const uint4*>(ar0 + k0 + half * 8);
        a0.q[1] = *reinterpret_cast<const uint4*>(ar0 + k0 + 16 + half * 8);
        a1.q[0] = *reinterpret_cast<const uint4*>(ar1 + k0 + half * 8);
        a1.q[1] = *reinterpret_cast<const uint4*>(ar1 + k0 + 16 + half * 8);
#pragma unroll
        for (int j = 0; j < 4; ++j) {
            const uint16_t* bp = br0 + (size_t)j * 16 * HH + k0 + half * 8;
            FragBF b;
            b.q[0] = *reinterpret_cast<const uint4*>(bp);
            b.q[1] = *reinterpret_cast<const uint4*>(bp + 16);
            acc[0][j] = __builtin_amdgcn_wmma_f32_16x16x32_bf16(
                false, a0.v, false, b.v, (short)0, acc[0][j], false, false);
            acc[1][j] = __builtin_amdgcn_wmma_f32_16x16x32_bf16(
                false, a1.v, false, b.v, (short)0, acc[1][j], false, false);
        }
    }

#pragma unroll
    for (int s = 0; s < 2; ++s) {
#pragma unroll
        for (int j = 0; j < 4; ++j) {
            int n = n0 + j * 16 + mh;
            float bb = bout[n];
#pragma unroll
            for (int r = 0; r < 8; ++r) {
                int m = m0 + s * 16 + half * 8 + r;      // m = t*B + b
                int t = m >> 6, b = m & 63;
                out[(size_t)((b << 6) + t) * VV + n] = acc[s][j][r] + bb;
            }
        }
    }
}

// ---------------------------------------------------------------------------
// Host: orchestrate the whole forward pass on `stream`.
// ---------------------------------------------------------------------------
extern "C" void kernel_launch(void* const* d_in, const int* in_sizes, int n_in,
                              void* d_out, int out_size, void* d_ws, size_t ws_size,
                              hipStream_t stream) {
    const int*   input_word = (const int*)  d_in[0];
    const int*   cond       = (const int*)  d_in[1];
    const float* emb_N      = (const float*)d_in[2];
    const float* Wih_N      = (const float*)d_in[3];
    const float* Whh_N      = (const float*)d_in[4];
    const float* bih_N      = (const float*)d_in[5];
    const float* bhh_N      = (const float*)d_in[6];
    const float* emb_D      = (const float*)d_in[7];
    const float* Wih_D      = (const float*)d_in[8];
    const float* Whh_D      = (const float*)d_in[9];
    const float* bih_D      = (const float*)d_in[10];
    const float* bhh_D      = (const float*)d_in[11];
    const float* emb_cond   = (const float*)d_in[12];
    const float* W_mean     = (const float*)d_in[13];
    const float* b_mean     = (const float*)d_in[14];
    const float* W_logvar   = (const float*)d_in[15];
    const float* b_logvar   = (const float*)d_in[16];
    const float* W_st       = (const float*)d_in[17];
    const float* b_st       = (const float*)d_in[18];
    const float* W_out      = (const float*)d_in[19];
    const float* b_out      = (const float*)d_in[20];
    const float* eps        = (const float*)d_in[21];
    float* out = (float*)d_out;

    // ---- workspace carve-up ----
    size_t off = 0;
    auto alloc = [&](size_t bytes) -> char* {
        off = (off + 255) & ~(size_t)255;
        char* p = (char*)d_ws + off;
        off += bytes;
        return p;
    };
    const size_t TBH = (size_t)TT * BB * HH;
    const size_t GH  = (size_t)GG * HH;
    const size_t BH  = (size_t)BB * HH;

    uint16_t* xe     = (uint16_t*)alloc(TBH * 2);
    uint16_t* xd     = (uint16_t*)alloc(TBH * 2);
    uint16_t* wihN   = (uint16_t*)alloc(GH * 2);
    uint16_t* whhN   = (uint16_t*)alloc(GH * 2);
    uint16_t* wihD   = (uint16_t*)alloc(GH * 2);
    uint16_t* whhD   = (uint16_t*)alloc(GH * 2);
    uint16_t* woutb  = (uint16_t*)alloc((size_t)VV * HH * 2);
    uint16_t* hsd    = (uint16_t*)alloc(TBH * 2);
    uint16_t* hA     = (uint16_t*)alloc(BH * 2);
    uint16_t* hB     = (uint16_t*)alloc(BH * 2);
    float*    cbuf   = (float*)   alloc(BH * 4);
    float*    hf32   = (float*)   alloc(BH * 4);
    float*    conde  = (float*)   alloc((size_t)BB * CDD * 4);
    float*    lat    = (float*)   alloc((size_t)BB * (ZZ + CDD) * 4);
    (void)ws_size; (void)n_in; (void)in_sizes; (void)out_size;

    // 1) weight conversion to bf16
    cvt_bf16_kernel<<<(int)(GH / 4 / 256), 256, 0, stream>>>(Wih_N, wihN, (int)(GH / 4));
    cvt_bf16_kernel<<<(int)(GH / 4 / 256), 256, 0, stream>>>(Whh_N, whhN, (int)(GH / 4));
    cvt_bf16_kernel<<<(int)(GH / 4 / 256), 256, 0, stream>>>(Wih_D, wihD, (int)(GH / 4));
    cvt_bf16_kernel<<<(int)(GH / 4 / 256), 256, 0, stream>>>(Whh_D, whhD, (int)(GH / 4));
    cvt_bf16_kernel<<<(int)((size_t)VV * HH / 4 / 256), 256, 0, stream>>>(W_out, woutb, (int)((size_t)VV * HH / 4));

    // 2) embedding gathers (enc + shifted dec tokens)
    embed_kernel<<<TT * BB, 256, 0, stream>>>(input_word, emb_N, emb_D, xe, xd);

    // 3) encoder init + scan
    enc_init_kernel<<<BB * HH / 256, 256, 0, stream>>>(cond, emb_cond, cbuf, hf32, hA, conde);
    for (int t = 0; t < TT; ++t) {
        uint16_t* hin  = (t & 1) ? hB : hA;
        uint16_t* hout = (t & 1) ? hA : hB;
        lstm_step_kernel<<<HH / 64, 512, 0, stream>>>(
            xe + (size_t)t * BB * HH, hin, wihN, whhN, bih_N, bhh_N,
            cbuf, hout, hf32, (uint16_t*)nullptr);
    }

    // 4) latent + decoder init
    latent_kernel<<<BB * ZZ / 256, 256, 0, stream>>>(hf32, W_mean, b_mean, W_logvar, b_logvar,
                                                     eps, conde, lat);
    dec_init_kernel<<<BB * HH / 256, 256, 0, stream>>>(lat, W_st, b_st, cbuf, hf32, hA);

    // 5) decoder scan (stores hs_d)
    for (int t = 0; t < TT; ++t) {
        uint16_t* hin  = (t & 1) ? hB : hA;
        uint16_t* hout = (t & 1) ? hA : hB;
        lstm_step_kernel<<<HH / 64, 512, 0, stream>>>(
            xd + (size_t)t * BB * HH, hin, wihD, whhD, bih_D, bhh_D,
            cbuf, hout, hf32, hsd + (size_t)t * BB * HH);
    }

    // 6) output projection -> [B,T,V]
    dim3 og(VV / 256, (TT * BB) / 32);
    out_gemm_kernel<<<og, 128, 0, stream>>>(hsd, woutb, b_out, out);
}
